// NVGPTAttention_72378788873099
// MI455X (gfx1250) — matrix operand order repaired and checked by reading, not compiled, not run on hardware
//
#include <hip/hip_runtime.h>

// ---------------------------------------------------------------------------
// NVGPT attention block for MI455X (gfx1250, wave32, WMMA bf16 + TDM path)
// ---------------------------------------------------------------------------
#define B_   4
#define S_   1024
#define HID_ 4096
#define NH_  32
#define HD_  128
#define K_   4096            // GEMM reduction dim (HID)
#define QKVN 12288           // 3*HID
#define SCALE_ 0.08838834764831845f   // HD^-0.5
#define NEG_INF_ (-1.0e30f)

typedef __attribute__((ext_vector_type(16))) __bf16 v16bf;
typedef __attribute__((ext_vector_type(8)))  __bf16 v8bf;
typedef __attribute__((ext_vector_type(4)))  __bf16 v4bf;
typedef __attribute__((ext_vector_type(8)))  float  v8f;
typedef __attribute__((ext_vector_type(4)))  unsigned int u32x4;
typedef __attribute__((ext_vector_type(8)))  int          i32x8;
typedef __attribute__((ext_vector_type(4)))  int          i32x4;

static __device__ __forceinline__ v8f wmma_bf16(v16bf a, v16bf b, v8f c) {
  // D = A(16x32 bf16) * B(32x16 bf16) + C(16x16 f32)
  return __builtin_amdgcn_wmma_f32_16x16x32_bf16(false, a, false, b,
                                                 (short)0, c, false, false);
}

static __device__ __forceinline__ v16bf mk16(v8bf lo, v8bf hi) {
  v16bf r;
#pragma unroll
  for (int i = 0; i < 8; ++i) { r[i] = lo[i]; r[i + 8] = hi[i]; }
  return r;
}

// A-fragment (16x32 bf16): lane l<16 -> row l, K chunks [0..7],[16..23];
// lane l>=16 -> row l-16, K chunks [8..15],[24..31].  (ISA 7.12.2)
static __device__ __forceinline__ v16bf load_fragA(const __bf16* base,
                                                   int lane, int stride) {
  int row = lane & 15;
  int kb  = (lane < 16) ? 0 : 8;
  const __bf16* p = base + (size_t)row * stride + kb;
  v8bf lo = *(const v8bf*)p;
  v8bf hi = *(const v8bf*)(p + 16);
  return mk16(lo, hi);
}

// ---------------------------------------------------------------------------
// Tensor Data Mover: DMA a [rows=128] x [cols=32 bf16] tile from a row-major
// tensor (row stride 4096 elems) into LDS with 4-DWORD padding after every
// 16 DWORDs (=> LDS row stride 40 bf16 = LDST).  D# per ISA ch.8.
// This toolchain exposes the 6-arg builtin:
//   (u32x4 g0, i32x8 g1, i32x4 g2, i32x4 g3, i32x8 g4, i32 cpol)
// ---------------------------------------------------------------------------
static __device__ __forceinline__ void tdm_load_tile(unsigned lds_byte_off,
                                                     const __bf16* gsrc,
                                                     unsigned n_rows_total) {
  unsigned long long ga = (unsigned long long)gsrc;
  u32x4 g0;
  g0[0] = 1u;                                   // count=1 valid descriptor
  g0[1] = lds_byte_off;                         // lds_addr  [63:32]
  g0[2] = (unsigned)(ga & 0xffffffffu);         // global_addr[95:64]
  g0[3] = (unsigned)((ga >> 32) & 0x01ffffffu)  // global_addr[120:96]
        | (2u << 30);                           // type=2 ("image") [127:126]
  i32x8 g1;
  g1[0] = (1 << 16)        // data_size = 1 -> 2 bytes (bf16)
        | (1 << 20)        // pad_enable
        | (3 << 22)        // pad_interval = 3 -> every 16 DWORDs (32 bf16)
        | (3 << 25);       // pad_amount   = 3 -> 4 DWORDs (8 bf16)
  g1[1] = (int)(4096u << 16);                   // tensor_dim0[15:0] = 4096
  g1[2] = (int)((n_rows_total & 0xffffu) << 16);// tensor_dim1[15:0]
  g1[3] = (int)((n_rows_total >> 16) & 0xffffu) // tensor_dim1[31:16]
        | (32 << 16);                           // tile_dim0 = 32 elems
  g1[4] = 128;                                  // tile_dim1 = 128 rows
  g1[5] = 4096;                                 // tensor_dim0_stride = 4096
  g1[6] = 0;
  g1[7] = 0;
  i32x4 gz4 = {0, 0, 0, 0};                     // 2D tile: groups 2/3 unused
  i32x8 gz8 = {0, 0, 0, 0, 0, 0, 0, 0};
  __builtin_amdgcn_tensor_load_to_lds(g0, g1, gz4, gz4, gz8, 0);
}

// ---------------------------------------------------------------------------
// fp32 -> bf16 bulk convert (vector of 4)
// ---------------------------------------------------------------------------
__global__ void cvt_bf16_kernel(const float* __restrict__ in,
                                __bf16* __restrict__ out, int n4) {
  int i = blockIdx.x * blockDim.x + threadIdx.x;
  if (i >= n4) return;
  float4 v = ((const float4*)in)[i];
  v4bf o;
  o[0] = (__bf16)v.x; o[1] = (__bf16)v.y; o[2] = (__bf16)v.z; o[3] = (__bf16)v.w;
  ((v4bf*)out)[i] = o;
}

// ---------------------------------------------------------------------------
// lora_mid[b,s,r] = sum_h hidden[b,s,h] * lora_in[b,r,h]   (rank 32, fp32)
// ---------------------------------------------------------------------------
__global__ __launch_bounds__(256)
void lora_mid_kernel(const float* __restrict__ hs,
                     const float* __restrict__ lin,
                     float* __restrict__ lm) {
  int t = blockIdx.x * 256 + threadIdx.x;   // 131072 threads
  int r = t & 31;
  int s = (t >> 5) & 1023;
  int b = t >> 15;
  const float4* hp = (const float4*)(hs + ((size_t)(b * S_ + s)) * HID_);
  const float4* lp = (const float4*)(lin + ((size_t)(b * 32 + r)) * HID_);
  float acc = 0.f;
  for (int i = 0; i < HID_ / 4; ++i) {
    float4 a = hp[i], c = lp[i];
    acc += a.x * c.x + a.y * c.y + a.z * c.z + a.w * c.w;
  }
  lm[t] = acc;
}

// ---------------------------------------------------------------------------
// Tiled bf16 WMMA GEMM: C[M,N] = A[M,K] * W[N,K]^T, M=4096, K=4096.
// Block: 256 thr (8 waves), tile 128x128.  A/B tiles are DMA'd into LDS by
// the Tensor Data Mover (wave0 -> A, wave1 -> B), double-buffered so the DMA
// of K-step k+1 overlaps the WMMA stream of step k.
// EPI=1: add rank-32 LoRA and store bf16 (QKV).  EPI=0: store fp32 (dense).
// ---------------------------------------------------------------------------
#define LDST 40   // padded LDS row stride (bf16 elems) = 32 + TDM pad of 8

template <int EPI>
__global__ __launch_bounds__(256)
void gemm_bf16_kernel(const __bf16* __restrict__ A,
                      const __bf16* __restrict__ W,
                      float* __restrict__ Cf, __bf16* __restrict__ Cb,
                      const float* __restrict__ loraMid,
                      const float* __restrict__ loraOut, int N) {
  __shared__ __bf16 As[2][128 * LDST];
  __shared__ __bf16 Bs[2][128 * LDST];

  const int n0  = blockIdx.x * 128;
  const int m0  = blockIdx.y * 128;
  const int tid = threadIdx.x;
  const int lane = tid & 31;
  const int w    = tid >> 5;
  const int wm   = w & 3;      // 4 wave-rows  -> rows wm*32 .. +31
  const int wn   = w >> 2;     // 2 wave-cols  -> cols wn*64 .. +63

  const unsigned ldsA[2] = {(unsigned)(size_t)&As[0][0],
                            (unsigned)(size_t)&As[1][0]};
  const unsigned ldsB[2] = {(unsigned)(size_t)&Bs[0][0],
                            (unsigned)(size_t)&Bs[1][0]};
  const __bf16* Abase = A + (size_t)m0 * K_;
  const __bf16* Wbase = W + (size_t)n0 * K_;

  v8f acc[2][4];
#pragma unroll
  for (int mt = 0; mt < 2; ++mt)
#pragma unroll
    for (int nt = 0; nt < 4; ++nt)
#pragma unroll
      for (int i = 0; i < 8; ++i) acc[mt][nt][i] = 0.f;

  // prime the pipeline: DMA K-slab 0 into buffer 0
  if (w == 0) tdm_load_tile(ldsA[0], Abase, 4096u);
  if (w == 1) tdm_load_tile(ldsB[0], Wbase, (unsigned)N);

  for (int it = 0; it < K_ / 32; ++it) {
    const int cur = it & 1;
    if (w < 2) __builtin_amdgcn_s_wait_tensorcnt(0);  // buf[cur] DMA done
    __syncthreads();  // publish buf[cur]; all waves done reading buf[cur^1]

    if (it + 1 < K_ / 32) {   // DMA next K-slab into the other buffer
      const size_t koff = (size_t)(it + 1) * 32;
      if (w == 0) tdm_load_tile(ldsA[cur ^ 1], Abase + koff, 4096u);
      if (w == 1) tdm_load_tile(ldsB[cur ^ 1], Wbase + koff, (unsigned)N);
    }

    v16bf af[2], bfr[4];
#pragma unroll
    for (int mt = 0; mt < 2; ++mt)
      af[mt] = load_fragA(&As[cur][(wm * 32 + mt * 16) * LDST], lane, LDST);
#pragma unroll
    for (int nt = 0; nt < 4; ++nt)
      bfr[nt] = load_fragA(&Bs[cur][(wn * 64 + nt * 16) * LDST], lane, LDST);

#pragma unroll
    for (int mt = 0; mt < 2; ++mt)
#pragma unroll
      for (int nt = 0; nt < 4; ++nt)
        acc[mt][nt] = wmma_bf16(af[mt], bfr[nt], acc[mt][nt]);
  }

  // --------------------------- epilogue ------------------------------------
  const int rhalf = (lane >> 4) & 1;   // C layout: M = r + 8*rhalf
#pragma unroll
  for (int mt = 0; mt < 2; ++mt) {
#pragma unroll
    for (int nt = 0; nt < 4; ++nt) {
      int col  = n0 + wn * 64 + nt * 16 + (lane & 15);
      int rowb = m0 + wm * 32 + mt * 16 + rhalf * 8;
      if (EPI == 1) {
        int b = rowb >> 10;   // rows of a block never cross a batch boundary
        const float* lo = loraOut + ((size_t)b * QKVN + col) * 32;
#pragma unroll
        for (int r = 0; r < 8; ++r) {
          int row = rowb + r;
          const float* lm = loraMid + (size_t)row * 32;
          float s = 0.f;
#pragma unroll
          for (int j = 0; j < 32; ++j) s += lm[j] * lo[j];
          Cb[(size_t)row * N + col] = (__bf16)(acc[mt][nt][r] + s);
        }
      } else {
#pragma unroll
        for (int r = 0; r < 8; ++r)
          Cf[(size_t)(rowb + r) * N + col] = acc[mt][nt][r];
      }
    }
  }
}

// ---------------------------------------------------------------------------
// RoPE in place on bf16 qkv (q and k halves), thread per (b,s,h,pair)
// ---------------------------------------------------------------------------
__global__ __launch_bounds__(256)
void rope_kernel(const int* __restrict__ pos, __bf16* __restrict__ qkv) {
  int t  = blockIdx.x * 256 + threadIdx.x;   // B*S*NH*64 threads
  int d  = t & 63;
  int hh = (t >> 6) & 31;
  int s  = (t >> 11) & 1023;
  int b  = t >> 21;
  float inv = __powf(10000.0f, -(float)d * (1.0f / 64.0f));
  float f   = (float)pos[s] * inv;
  float c   = __cosf(f), sn = __sinf(f);
  __bf16* q = qkv + ((size_t)(b * S_ + s)) * QKVN + hh * HD_;
  float x1 = (float)q[d], x2 = (float)q[d + 64];
  q[d]      = (__bf16)(x1 * c - x2 * sn);
  q[d + 64] = (__bf16)(x2 * c + x1 * sn);
  __bf16* k = q + HID_;
  x1 = (float)k[d]; x2 = (float)k[d + 64];
  k[d]      = (__bf16)(x1 * c - x2 * sn);
  k[d + 64] = (__bf16)(x2 * c + x1 * sn);
}

// ---------------------------------------------------------------------------
// Fused causal flash attention. One wave32 owns a 16-row Q tile of one head.
// Streams K/V in 32-row chunks, online softmax in fp32 C-registers, P is
// transposed C-layout -> A-layout through a per-wave LDS bounce.
// ---------------------------------------------------------------------------
__global__ __launch_bounds__(256)
void attn_kernel(const __bf16* __restrict__ qkv, __bf16* __restrict__ attn) {
  __shared__ __bf16 plds[8][16 * LDST];   // per-wave P tile (16x32 padded)

  const int lane = threadIdx.x & 31;
  const int w    = threadIdx.x >> 5;
  const int g    = blockIdx.x * 8 + w;    // wave-tile id
  const int qt   = g & 63;
  const int h    = (g >> 6) & 31;
  const int b    = g >> 11;
  const int q0   = qt * 16;

  const __bf16* qbase = qkv + (size_t)b * S_ * QKVN + (size_t)h * HD_;
  const __bf16* kbase = qbase + HID_;
  const __bf16* vbase = qbase + 2 * HID_;

  const int rhalf = (lane >> 4) & 1;

  // Q fragments: 16x128 bf16 as 4 A-frags, held for the whole tile
  v16bf qf[4];
#pragma unroll
  for (int kb = 0; kb < 4; ++kb)
    qf[kb] = load_fragA(qbase + (size_t)q0 * QKVN + kb * 32, lane, QKVN);

  float m_run[8], l_run[8];
  v8f o[8];
#pragma unroll
  for (int r = 0; r < 8; ++r) { m_run[r] = NEG_INF_; l_run[r] = 0.f; }
#pragma unroll
  for (int c = 0; c < 8; ++c)
#pragma unroll
    for (int i = 0; i < 8; ++i) o[c][i] = 0.f;

  const int nchunks = qt / 2 + 1;   // 32-row K/V chunks covering cols <= q0+15
  for (int jc = 0; jc < nchunks; ++jc) {
    const int kr0 = jc * 32;

    // ---- S = Q @ K^T  (two 16x16 score tiles) ----
    v8f s0, s1;
#pragma unroll
    for (int i = 0; i < 8; ++i) { s0[i] = 0.f; s1[i] = 0.f; }
#pragma unroll
    for (int kb = 0; kb < 4; ++kb) {
      const __bf16* p0 = kbase + (size_t)kr0 * QKVN + kb * 32;
      v16bf b0 = load_fragA(p0, lane, QKVN);
      v16bf b1 = load_fragA(p0 + (size_t)16 * QKVN, lane, QKVN);
      s0 = wmma_bf16(qf[kb], b0, s0);
      s1 = wmma_bf16(qf[kb], b1, s1);
    }

    // ---- scale + causal mask + online softmax stats ----
    float corr[8], e0a[8], e1a[8];
#pragma unroll
    for (int r = 0; r < 8; ++r) {
      int rowg = q0 + r + rhalf * 8;
      float a0 = s0[r] * SCALE_;
      float a1 = s1[r] * SCALE_;
      if (kr0 + (lane & 15) > rowg)      a0 = NEG_INF_;
      if (kr0 + 16 + (lane & 15) > rowg) a1 = NEG_INF_;
      float mx = fmaxf(a0, a1);
#pragma unroll
      for (int d = 1; d < 16; d <<= 1) mx = fmaxf(mx, __shfl_xor(mx, d, 32));
      float mn = fmaxf(m_run[r], mx);
      float cr = __expf(m_run[r] - mn);
      m_run[r] = mn;
      float e0 = __expf(a0 - mn), e1 = __expf(a1 - mn);
      float sm = e0 + e1;
#pragma unroll
      for (int d = 1; d < 16; d <<= 1) sm += __shfl_xor(sm, d, 32);
      l_run[r] = l_run[r] * cr + sm;
      corr[r] = cr; e0a[r] = e0; e1a[r] = e1;
    }
#pragma unroll
    for (int c = 0; c < 8; ++c)
#pragma unroll
      for (int r = 0; r < 8; ++r) o[c][r] *= corr[r];

    // ---- P: C-layout -> LDS -> A-layout fragment ----
    __bf16* pl = plds[w];
#pragma unroll
    for (int r = 0; r < 8; ++r) {
      int pr = r + rhalf * 8;
      pl[pr * LDST + (lane & 15)]      = (__bf16)e0a[r];
      pl[pr * LDST + 16 + (lane & 15)] = (__bf16)e1a[r];
    }
    asm volatile("s_wait_dscnt 0" ::: "memory");
    v16bf pf = load_fragA(pl, lane, LDST);

    // ---- O += P @ V  (8 N-chunks of 16 over HD=128) ----
#pragma unroll
    for (int c = 0; c < 8; ++c) {
      const __bf16* vp = vbase + (size_t)(kr0 + (lane >> 4) * 8) * QKVN + c * 16 + (lane & 15);
      v16bf vf;
#pragma unroll
      for (int i = 0; i < 8; ++i) { vf[i] = vp[(size_t)i * QKVN]; vf[i + 8] = vp[(size_t)(i + 16) * QKVN]; }
      o[c] = wmma_bf16(pf, vf, o[c]);
    }
  }

  // ---- finalize: O / l, store bf16 into attn[b,s,h*HD+d] ----
  float inv[8];
#pragma unroll
  for (int r = 0; r < 8; ++r) inv[r] = 1.0f / l_run[r];
#pragma unroll
  for (int c = 0; c < 8; ++c)
#pragma unroll
    for (int r = 0; r < 8; ++r) {
      int row = q0 + r + rhalf * 8;
      attn[((size_t)(b * S_ + row)) * HID_ + h * HD_ + c * 16 + (lane & 15)] =
          (__bf16)(o[c][r] * inv[r]);
    }
}

// ---------------------------------------------------------------------------
// Host-side launch
// ---------------------------------------------------------------------------
extern "C" void kernel_launch(void* const* d_in, const int* in_sizes, int n_in,
                              void* d_out, int out_size, void* d_ws,
                              size_t ws_size, hipStream_t stream) {
  (void)in_sizes; (void)n_in; (void)out_size; (void)ws_size;
  const int*   positions = (const int*)d_in[0];
  const float* hidden    = (const float*)d_in[1];
  const float* w_qkv     = (const float*)d_in[2];
  const float* w_dense   = (const float*)d_in[3];
  const float* lora_in   = (const float*)d_in[4];
  const float* lora_out  = (const float*)d_in[5];
  float*       out       = (float*)d_out;

  char* ws = (char*)d_ws;
  const size_t nHS   = (size_t)B_ * S_ * HID_;        // 16.7M
  const size_t nWQKV = (size_t)QKVN * HID_;           // 50.3M
  const size_t nWD   = (size_t)HID_ * HID_;           // 16.7M
  size_t off = 0;
  __bf16* hs_bf   = (__bf16*)(ws + off); off += nHS * 2;
  __bf16* wqkv_bf = (__bf16*)(ws + off); off += nWQKV * 2;
  __bf16* wd_bf   = (__bf16*)(ws + off); off += nWD * 2;
  float*  lmid    = (float*)(ws + off);  off += (size_t)B_ * S_ * 32 * 4;
  __bf16* qkv_bf  = (__bf16*)(ws + off); off += (size_t)B_ * S_ * QKVN * 2;
  __bf16* attn_bf = (__bf16*)(ws + off); off += nHS * 2;

  // 1) fp32 -> bf16 converts
  cvt_bf16_kernel<<<(int)(nHS / 4 / 256), 256, 0, stream>>>(hidden, hs_bf,
                                                            (int)(nHS / 4));
  cvt_bf16_kernel<<<(int)(nWQKV / 4 / 256), 256, 0, stream>>>(w_qkv, wqkv_bf,
                                                              (int)(nWQKV / 4));
  cvt_bf16_kernel<<<(int)(nWD / 4 / 256), 256, 0, stream>>>(w_dense, wd_bf,
                                                            (int)(nWD / 4));
  // 2) LoRA mid projection (rank 32, fp32)
  lora_mid_kernel<<<512, 256, 0, stream>>>(hidden, lora_in, lmid);

  // 3) QKV GEMM (TDM-fed) + LoRA epilogue -> bf16 qkv
  gemm_bf16_kernel<1><<<dim3(QKVN / 128, (B_ * S_) / 128), 256, 0, stream>>>(
      hs_bf, wqkv_bf, nullptr, qkv_bf, lmid, lora_out, QKVN);

  // 4) RoPE on q and k, in place
  rope_kernel<<<(B_ * S_ * NH_ * 64) / 256, 256, 0, stream>>>(positions, qkv_bf);

  // 5) fused causal flash attention -> bf16 attn
  attn_kernel<<<(B_ * NH_ * (S_ / 16)) / 8, 256, 0, stream>>>(qkv_bf, attn_bf);

  // 6) output projection (TDM-fed) -> fp32 d_out
  gemm_bf16_kernel<0><<<dim3(HID_ / 128, (B_ * S_) / 128), 256, 0, stream>>>(
      attn_bf, wd_bf, out, nullptr, nullptr, nullptr, HID_);
}
